// Attention_1279900254845
// MI455X (gfx1250) — compile-verified, matrix operand order checked by reading, not run on hardware
//
#include <hip/hip_runtime.h>

typedef _Float16 f16;
typedef __attribute__((ext_vector_type(8)))  _Float16 v8h;
typedef __attribute__((ext_vector_type(16))) _Float16 v16h;
typedef __attribute__((ext_vector_type(8)))  float    v8f;

constexpr int kS  = 256;   // tokens (Q = K)
constexpr int kC  = 128;   // input channels (CQ=CK=CV)
constexpr int kH  = 8;     // heads
constexpr int kCH = 32;    // head dim
constexpr int kHC = 256;   // H*CH
constexpr int kNR = 256;   // rows

// padded LDS row strides (halves): odd multiples of 8 -> 16B-aligned b128
// loads with lanes spread across banks
constexpr int kLdX = 136;  // 128 + 8
constexpr int kLdW = 136;  // weight^T rows (K=128)
constexpr int kLdQ = 40;   // 32 + 8 (Q/K tiles, K-dim 32)
constexpr int kLdP = 264;  // 256 + 8 (P and V^T rows, K-dim 256)

// ---- fragment loader (CDNA5 16-bit WMMA layout, ISA 7.12.2) ----
// Source is [M][K] (A) or [N][K] (B^T staged): lane owns row (base + lane&15),
// element i holds K = k0 + 8*(lane>>4) + i + (i>=8 ? 8 : 0)  -> two
// contiguous 16-byte runs -> exactly two ds_load_b128 per fragment.
__device__ __forceinline__ v16h ld_frag(const f16* p, int row0, int ld,
                                        int k0, int lane) {
  const f16* q = p + (size_t)(row0 + (lane & 15)) * ld + k0 + ((lane >> 4) << 3);
  v8h lo = *(const v8h*)(q);
  v8h hi = *(const v8h*)(q + 16);
  return __builtin_shufflevector(lo, hi, 0, 1, 2, 3, 4, 5, 6, 7,
                                 8, 9, 10, 11, 12, 13, 14, 15);
}

__device__ __forceinline__ v8f wmma_f16(v16h a, v16h b, v8f c) {
  return __builtin_amdgcn_wmma_f32_16x16x32_f16(false, a, false, b,
                                                (short)0, c, false, false);
}

// scheduling fence: keep register pressure bounded (no hoisting across)
__device__ __forceinline__ void sched_fence() {
  __builtin_amdgcn_sched_barrier(0);
}

// ---- CDNA5 async global->LDS copy (ASYNCcnt tracked, no VGPR data) ----
__device__ __forceinline__ void async_ld_b128(uint32_t lds_byte_off, const f16* g) {
  asm volatile("global_load_async_to_lds_b128 %0, %1, off"
               :: "v"(lds_byte_off), "v"(g) : "memory");
}
__device__ __forceinline__ void wait_async0() {
  asm volatile("s_wait_asynccnt 0x0" ::: "memory");
}
__device__ __forceinline__ uint32_t lds_off(const void* p) {
  return (uint32_t)(uintptr_t)p;   // low 32 bits of shared aperture = LDS offset
}

// ===================== Kernel 1: input projections =====================
// Templated on W (0=wq,1=wk,2=wv,3=wg): fully specialized epilogues.
template <int W>
__global__ __launch_bounds__(256, 1)
void proj_kernel(const float* __restrict__ xg,
                 const float* __restrict__ wp,
                 const float* __restrict__ bg,
                 f16* __restrict__ dst) {
  extern __shared__ f16 smem[];
  f16* xs = smem;                 // [64][kLdX]  x-tile
  f16* wt = smem + 64 * kLdX;     // [256][kLdW] weight^T  (wt[n][k])
  const int nr  = blockIdx.x >> 2;
  const int s0  = (blockIdx.x & 3) * 64;
  const int tid = threadIdx.x;
  const int lane = tid & 31, wave = tid >> 5, hg = lane >> 4;

  const size_t xbase = ((size_t)nr * kS + s0) * kC;   // 64 rows contiguous
  for (int i = tid; i < 64 * kC; i += 256)
    xs[(i >> 7) * kLdX + (i & 127)] = (f16)xg[xbase + i];
  for (int i = tid; i < kC * kHC; i += 256) {         // wp[k][n] -> wt[n][k]
    int k = i >> 8, n = i & 255;
    wt[n * kLdW + k] = (f16)wp[i];
  }
  __syncthreads();

  const float scale = 0.17677669529663687f;  // 1/sqrt(32)
#pragma unroll 1
  for (int j = 0; j < 8; ++j) {              // 8 waves x 8 tiles = 64 tiles
    int t  = wave * 8 + j;
    int m0 = (t >> 4) * 16;
    int n0 = (t & 15) * 16;
    // batch all fragments, then chain WMMAs (hide LDS latency)
    v16h a0 = ld_frag(xs, m0, kLdX, 0,  lane);
    v16h a1 = ld_frag(xs, m0, kLdX, 32, lane);
    v16h a2 = ld_frag(xs, m0, kLdX, 64, lane);
    v16h a3 = ld_frag(xs, m0, kLdX, 96, lane);
    v16h b0 = ld_frag(wt, n0, kLdW, 0,  lane);
    v16h b1 = ld_frag(wt, n0, kLdW, 32, lane);
    v16h b2 = ld_frag(wt, n0, kLdW, 64, lane);
    v16h b3 = ld_frag(wt, n0, kLdW, 96, lane);
    v8f acc = {};
    acc = wmma_f16(a0, b0, acc);
    acc = wmma_f16(a1, b1, acc);
    acc = wmma_f16(a2, b2, acc);
    acc = wmma_f16(a3, b3, acc);

    int n = n0 + (lane & 15);
    if (W == 3) {                    // gate: sigmoid(x@wg + bg), [NR][S][HC]
      float bgv = bg[n];
#pragma unroll
      for (int r = 0; r < 8; ++r) {
        int srow = s0 + m0 + r + (hg << 3);
        float gv = __builtin_amdgcn_rcpf(1.0f + __expf(-(acc[r] + bgv)));
        dst[((size_t)nr * kS + srow) * kHC + n] = (f16)gv;
      }
    } else {                         // Q/K/V: head-split [NR][H][S][CH]
      const float mul = (W == 0) ? scale : 1.0f;
      int hh = n >> 5, ch = n & 31;
#pragma unroll
      for (int r = 0; r < 8; ++r) {
        int srow = s0 + m0 + r + (hg << 3);
        dst[(((size_t)nr * kH + hh) * kS + srow) * kCH + ch] = (f16)(acc[r] * mul);
      }
    }
    sched_fence();
  }
}

// ===================== Kernel 2: attention (one block per nr,h) =====================
__global__ __launch_bounds__(256, 1)
void attn_kernel(const float* __restrict__ bias_mask,
                 const float* __restrict__ bias_pair,
                 const f16* __restrict__ Qh, const f16* __restrict__ Kh,
                 const f16* __restrict__ Vh, const f16* __restrict__ Gh,
                 f16* __restrict__ Oh) {
  extern __shared__ f16 smem[];
  f16* Qs = smem;                          // [256][kLdQ]
  f16* Ks = Qs + kS * kLdQ;                // [256][kLdQ]
  f16* Vt = Ks + kS * kLdQ;                // [32][kLdP]  V^T: Vt[ch][key]
  f16* Ps = Vt + kCH * kLdP;               // [256][kLdP] probabilities
  const int nr  = blockIdx.x >> 3;
  const int h   = blockIdx.x & 7;
  const int tid = threadIdx.x;
  const int lane = tid & 31, wave = tid >> 5, hg = lane >> 4;

  __builtin_prefetch(bias_pair + ((size_t)h * kS) * kS, 0, 0);  // global_prefetch_b8

  const size_t base = ((size_t)nr * kH + h) * kS * kCH;  // in halves
  // async 16B-chunk copies into padded rows: row = 32 halves = 4 chunks
  const uint32_t qs_off = lds_off(Qs), ks_off = lds_off(Ks);
  for (int i = tid; i < kS * kCH / 8; i += 256) {     // 1024 chunks each
    int key = i >> 2, c8 = (i & 3) * 8;               // chunk start (halves)
    uint32_t doff = (uint32_t)((key * kLdQ + c8) * 2);
    const f16* qsrc = Qh + base + key * kCH + c8;
    const f16* ksrc = Kh + base + key * kCH + c8;
    async_ld_b128(qs_off + doff, qsrc);
    async_ld_b128(ks_off + doff, ksrc);
  }
  const f16* vsrc = Vh + base;
  for (int i = tid; i < kS * kCH; i += 256) {         // transpose V -> Vt[ch][key]
    int key = i >> 5, ch = i & 31;
    Vt[ch * kLdP + key] = vsrc[i];
  }
  wait_async0();
  __syncthreads();

  // --- scores + softmax; each wave owns 16 query rows per iteration ---
#pragma unroll 1
  for (int it = 0; it < 2; ++it) {
    int qt = wave + it * 8;
    int q0 = qt * 16;
    v16h a = ld_frag(Qs, q0, kLdQ, 0, lane);   // Q fragment, K-dim = 32
    v8f sacc[16];
#pragma unroll
    for (int g = 0; g < 4; ++g) {              // 4 B-frags per fenced group
      v16h b0 = ld_frag(Ks, (g * 4 + 0) * 16, kLdQ, 0, lane);
      v16h b1 = ld_frag(Ks, (g * 4 + 1) * 16, kLdQ, 0, lane);
      v16h b2 = ld_frag(Ks, (g * 4 + 2) * 16, kLdQ, 0, lane);
      v16h b3 = ld_frag(Ks, (g * 4 + 3) * 16, kLdQ, 0, lane);
      v8f z0 = {}, z1 = {}, z2 = {}, z3 = {};
      sacc[g * 4 + 0] = wmma_f16(a, b0, z0);
      sacc[g * 4 + 1] = wmma_f16(a, b1, z1);
      sacc[g * 4 + 2] = wmma_f16(a, b2, z2);
      sacc[g * 4 + 3] = wmma_f16(a, b3, z3);
      sched_fence();   // cap live B-fragments: no cross-group hoisting
    }
    // additive biases (D-fragment layout: row = q0+r+8*hg, col = lane&15)
#pragma unroll
    for (int g = 0; g < 4; ++g) {
#pragma unroll
      for (int nt4 = 0; nt4 < 4; ++nt4) {
        int nt = g * 4 + nt4;
        int n = nt * 16 + (lane & 15);
        float bm = bias_mask[(size_t)nr * kS + n];
#pragma unroll
        for (int r = 0; r < 8; ++r) {
          int q = q0 + r + (hg << 3);
          sacc[nt][r] += bm + bias_pair[((size_t)h * kS + q) * kS + n];
        }
      }
      sched_fence();   // cap in-flight bias loads
    }
    // softmax over 256 keys: 16 in-lane values + 16-lane shuffle reduction
    float rinv[8];
#pragma unroll
    for (int r = 0; r < 8; ++r) {
      float m = sacc[0][r];
#pragma unroll
      for (int nt = 1; nt < 16; ++nt) m = fmaxf(m, sacc[nt][r]);
#pragma unroll
      for (int off = 8; off >= 1; off >>= 1) m = fmaxf(m, __shfl_xor(m, off, 32));
      float s = 0.0f;
#pragma unroll
      for (int nt = 0; nt < 16; ++nt) {
        float e = __expf(sacc[nt][r] - m);
        sacc[nt][r] = e;
        s += e;
      }
#pragma unroll
      for (int off = 8; off >= 1; off >>= 1) s += __shfl_xor(s, off, 32);
      rinv[r] = __builtin_amdgcn_rcpf(s);
    }
#pragma unroll
    for (int nt = 0; nt < 16; ++nt) {
      int n = nt * 16 + (lane & 15);
#pragma unroll
      for (int r = 0; r < 8; ++r) {
        int q = q0 + r + (hg << 3);
        Ps[(size_t)q * kLdP + n] = (f16)(sacc[nt][r] * rinv[r]);
      }
    }
    sched_fence();
  }
  __syncthreads();

  // --- O = P @ V, gated epilogue ---
#pragma unroll 1
  for (int j = 0; j < 4; ++j) {
    int t  = wave + j * 8;        // 32 tiles: 16 q-tiles x 2 ch-tiles
    int qt = t & 15, ct = t >> 4;
    int q0 = qt * 16, c0 = ct * 16;
    v8f acc = {};
#pragma unroll
    for (int g = 0; g < 2; ++g) {      // 4 (A,B) pairs per fenced group
      int k0 = g * 128;
      v16h a0 = ld_frag(Ps, q0, kLdP, k0 + 0,  lane);
      v16h a1 = ld_frag(Ps, q0, kLdP, k0 + 32, lane);
      v16h a2 = ld_frag(Ps, q0, kLdP, k0 + 64, lane);
      v16h a3 = ld_frag(Ps, q0, kLdP, k0 + 96, lane);
      v16h b0 = ld_frag(Vt, c0, kLdP, k0 + 0,  lane);
      v16h b1 = ld_frag(Vt, c0, kLdP, k0 + 32, lane);
      v16h b2 = ld_frag(Vt, c0, kLdP, k0 + 64, lane);
      v16h b3 = ld_frag(Vt, c0, kLdP, k0 + 96, lane);
      acc = wmma_f16(a0, b0, acc);
      acc = wmma_f16(a1, b1, acc);
      acc = wmma_f16(a2, b2, acc);
      acc = wmma_f16(a3, b3, acc);
      sched_fence();
    }
    int ch = c0 + (lane & 15);
    int n  = h * kCH + ch;
#pragma unroll
    for (int r = 0; r < 8; ++r) {
      int q = q0 + r + (hg << 3);
      size_t gi = ((size_t)nr * kS + q) * kHC + n;
      float g = (float)Gh[gi];
      Oh[gi] = (f16)(acc[r] * g);
    }
    sched_fence();
  }
}

// ===================== Kernel 3: output projection =====================
__global__ __launch_bounds__(256, 1)
void outproj_kernel(const f16* __restrict__ Oh,
                    const float* __restrict__ wo,
                    const float* __restrict__ bo,
                    float* __restrict__ out) {
  extern __shared__ f16 smem[];
  f16* As = smem;                 // [64][kLdP]  gated attention rows
  f16* wt = smem + 64 * kLdP;     // [128][kLdP] wo^T: wt[n][k]
  const int nr  = blockIdx.x >> 2;
  const int s0  = (blockIdx.x & 3) * 64;
  const int tid = threadIdx.x;
  const int lane = tid & 31, wave = tid >> 5, hg = lane >> 4;

  // async 16B-chunk copy of the 64x256 f16 tile into padded rows
  const f16* asrc = Oh + ((size_t)nr * kS + s0) * kHC;
  const uint32_t as_off = lds_off(As);
  for (int i = tid; i < 64 * kHC / 8; i += 256) {   // 2048 chunks
    int row = i >> 5, c8 = (i & 31) * 8;
    async_ld_b128(as_off + (uint32_t)((row * kLdP + c8) * 2),
                  asrc + row * kHC + c8);
  }
  for (int i = tid; i < kHC * kC; i += 256) {       // wo[k][n] -> wt[n][k]
    int k = i >> 7, n = i & 127;
    wt[n * kLdP + k] = (f16)wo[i];
  }
  wait_async0();
  __syncthreads();

#pragma unroll 1
  for (int j = 0; j < 4; ++j) {
    int t  = wave + j * 8;           // 32 tiles: 4 m x 8 n
    int m0 = (t >> 3) * 16, n0 = (t & 7) * 16;
    v8f acc = {};
#pragma unroll
    for (int g = 0; g < 2; ++g) {    // 4 (A,B) pairs per fenced group
      int k0 = g * 128;
      v16h a0 = ld_frag(As, m0, kLdP, k0 + 0,  lane);
      v16h a1 = ld_frag(As, m0, kLdP, k0 + 32, lane);
      v16h a2 = ld_frag(As, m0, kLdP, k0 + 64, lane);
      v16h a3 = ld_frag(As, m0, kLdP, k0 + 96, lane);
      v16h b0 = ld_frag(wt, n0, kLdP, k0 + 0,  lane);
      v16h b1 = ld_frag(wt, n0, kLdP, k0 + 32, lane);
      v16h b2 = ld_frag(wt, n0, kLdP, k0 + 64, lane);
      v16h b3 = ld_frag(wt, n0, kLdP, k0 + 96, lane);
      acc = wmma_f16(a0, b0, acc);
      acc = wmma_f16(a1, b1, acc);
      acc = wmma_f16(a2, b2, acc);
      acc = wmma_f16(a3, b3, acc);
      sched_fence();
    }
    int c = n0 + (lane & 15);
    float bias = bo[c];
#pragma unroll
    for (int r = 0; r < 8; ++r) {
      int srow = s0 + m0 + r + (hg << 3);
      out[((size_t)nr * kS + srow) * kC + c] = acc[r] + bias;
    }
    sched_fence();
  }
}

extern "C" void kernel_launch(void* const* d_in, const int* in_sizes, int n_in,
                              void* d_out, int out_size, void* d_ws, size_t ws_size,
                              hipStream_t stream) {
  const float* q_x       = (const float*)d_in[0];
  const float* kv_x      = (const float*)d_in[1];
  const float* bias_mask = (const float*)d_in[2];
  const float* bias_pair = (const float*)d_in[3];
  const float* wq = (const float*)d_in[4];
  const float* wk = (const float*)d_in[5];
  const float* wv = (const float*)d_in[6];
  const float* wg = (const float*)d_in[7];
  const float* bg = (const float*)d_in[8];
  const float* wo = (const float*)d_in[9];
  const float* bo = (const float*)d_in[10];
  float* out = (float*)d_out;

  char* wsb = (char*)d_ws;
  const size_t SZ = (size_t)kNR * kH * kS * kCH * sizeof(f16);  // 33.5 MB each
  f16* Qh = (f16*)(wsb + 0 * SZ);   // [NR][H][S][CH], pre-scaled
  f16* Kh = (f16*)(wsb + 1 * SZ);   // [NR][H][S][CH]
  f16* Vh = (f16*)(wsb + 2 * SZ);   // [NR][H][S][CH]
  f16* Gh = (f16*)(wsb + 3 * SZ);   // [NR][S][HC] sigmoid gate
  f16* Oh = (f16*)(wsb + 4 * SZ);   // [NR][S][HC] gated attention out

  dim3 blk(256);
  const size_t smem1 = (64 * kLdX + kHC * kLdW) * sizeof(f16);
  proj_kernel<0><<<dim3(kNR * 4), blk, smem1, stream>>>(q_x,  wq, bg, Qh);
  proj_kernel<1><<<dim3(kNR * 4), blk, smem1, stream>>>(kv_x, wk, bg, Kh);
  proj_kernel<2><<<dim3(kNR * 4), blk, smem1, stream>>>(kv_x, wv, bg, Vh);
  proj_kernel<3><<<dim3(kNR * 4), blk, smem1, stream>>>(q_x,  wg, bg, Gh);
  attn_kernel<<<dim3(kNR * kH), blk,
                (2 * kS * kLdQ + kCH * kLdP + kS * kLdP) * sizeof(f16), stream>>>(
      bias_mask, bias_pair, Qh, Kh, Vh, Gh, Oh);
  outproj_kernel<<<dim3(kNR * 4), blk,
                   (64 * kLdP + kC * kLdP) * sizeof(f16), stream>>>(
      Oh, wo, bo, out);
}
